// BatchedDiffPool_8753143349922
// MI455X (gfx1250) — matrix operand order repaired and checked by reading, not compiled, not run on hardware
//
#include <hip/hip_runtime.h>
#include <hip/hip_bf16.h>

// ---------------- problem constants ----------------
#define N_TOT 16384
#define Fdim  128
#define Kc    64
#define Bg    32
#define NPG   512              // nodes per graph (N_TOT / Bg)
#define KB_TOT 2048            // Kc * Bg
#define LDK   520              // padded LDS row stride (f16): 512 + 8 -> distinct banks, 16B aligned

// flat float offsets into d_out (return order: Zp, Ap, seg_ids, S_local, Sbd)
#define ZP_OFF  ((size_t)0)
#define AP_OFF  ((size_t)262144)
#define SEG_OFF ((size_t)4456448)
#define SL_OFF  ((size_t)4458496)
#define SBD_OFF ((size_t)5507072)

typedef __attribute__((ext_vector_type(16))) _Float16 v16h;
typedef __attribute__((ext_vector_type(8)))  _Float16 v8h;
typedef __attribute__((ext_vector_type(8)))  float    v8f;

// ---------------- WMMA fragment helpers (ISA 7.12.2 layouts, wave32) ----------------
// All LDS matrices are stored K-contiguous ("transposed"): X_T[n or m][k], row stride LDK.

// A fragment 16x32 f16 from row-major f32 global, element A[m0+m][k0+k].
// Lane L<16: M=L, K {0..7}u{16..23}; lane L>=16: M=L-16, K {8..15}u{24..31}.
// Vectorized: 4 x global_load_b128 per lane.
__device__ inline v16h load_a_frag_global(const float* Ab, int m0, int k0, int ld) {
  int lane = threadIdx.x & 31;
  const float* rp = Ab + (size_t)(m0 + (lane & 15)) * ld + k0 + ((lane >> 4) & 1) * 8;
  float4 q0 = ((const float4*)rp)[0];
  float4 q1 = ((const float4*)rp)[1];
  float4 q2 = ((const float4*)(rp + 16))[0];
  float4 q3 = ((const float4*)(rp + 16))[1];
  v16h a;
  a[0]=(_Float16)q0.x;  a[1]=(_Float16)q0.y;  a[2]=(_Float16)q0.z;  a[3]=(_Float16)q0.w;
  a[4]=(_Float16)q1.x;  a[5]=(_Float16)q1.y;  a[6]=(_Float16)q1.z;  a[7]=(_Float16)q1.w;
  a[8]=(_Float16)q2.x;  a[9]=(_Float16)q2.y;  a[10]=(_Float16)q2.z; a[11]=(_Float16)q2.w;
  a[12]=(_Float16)q3.x; a[13]=(_Float16)q3.y; a[14]=(_Float16)q3.z; a[15]=(_Float16)q3.w;
  return a;
}

// A fragment 16x32 where A[m][k] = ST[m][k] (ST is K-contiguous LDS, stride LDK).
// Two contiguous 8-f16 runs -> 2 x ds_load_b128.
__device__ inline v16h load_at_frag_ldsT(const _Float16* ST, int m0, int k0) {
  int lane = threadIdx.x & 31;
  const _Float16* rp = ST + (size_t)(m0 + (lane & 15)) * LDK + k0 + ((lane >> 4) & 1) * 8;
  v8h lo = *(const v8h*)rp;
  v8h hi = *(const v8h*)(rp + 16);
  v16h a;
  #pragma unroll
  for (int i = 0; i < 8; ++i) { a[i] = lo[i]; a[8 + i] = hi[i]; }
  return a;
}

// B fragment 32x16 where B[k][n] = BT[n][k] (K-contiguous LDS, stride LDK).
// Lane n<16: N=n, K {0..7}u{16..23}; lane n+16: K {8..15}u{24..31} -> two 8-f16 runs, 2 x ds_load_b128.
__device__ inline v16h load_b_frag_ldsT(const _Float16* BT, int k0, int n0) {
  int lane = threadIdx.x & 31;
  const _Float16* rp = BT + (size_t)(n0 + (lane & 15)) * LDK + k0 + ((lane >> 4) & 1) * 8;
  v8h lo = *(const v8h*)rp;
  v8h hi = *(const v8h*)(rp + 16);
  v16h b;
  #pragma unroll
  for (int i = 0; i < 8; ++i) { b[i] = lo[i]; b[8 + i] = hi[i]; }
  return b;
}

// C/D 16x16 f32 -> global f32 row-major: VGPR r -> M = r or 8+r, N = lane&15
__device__ inline void store_c_frag_global(float* D, int m0, int n0, int ld, v8f c) {
  int lane = threadIdx.x & 31;
  int n  = n0 + (lane & 15);
  int mb = m0 + ((lane >> 4) & 1) * 8;
  #pragma unroll
  for (int r = 0; r < 8; ++r) D[(size_t)(mb + r) * ld + n] = c[r];
}

// C/D 16x16 f32 -> K-contiguous f16 LDS (DT[n][m], stride LDK): 8 contiguous f16 = 1 ds_store_b128
__device__ inline void store_c_frag_ldsT(_Float16* DT, int m0, int n0, v8f c) {
  int lane = threadIdx.x & 31;
  int n  = n0 + (lane & 15);
  int mb = m0 + ((lane >> 4) & 1) * 8;
  v8h h;
  #pragma unroll
  for (int r = 0; r < 8; ++r) h[r] = (_Float16)c[r];
  *(v8h*)(DT + (size_t)n * LDK + mb) = h;
}

// ---------------- kernel 1: zero-fill Sbd, write seg_ids ----------------
__global__ void fill_kernel(float* out) {
  size_t i = (size_t)blockIdx.x * blockDim.x + threadIdx.x;
  float4* p = (float4*)(out + SBD_OFF);
  if (i < ((size_t)N_TOT * KB_TOT) / 4) {
    float4 z; z.x = 0.f; z.y = 0.f; z.z = 0.f; z.w = 0.f;
    p[i] = z;
  }
  if (i < KB_TOT) out[SEG_OFF + i] = (float)((int)i >> 6);
}

// ---------------- kernel 2: S_local = softmax(Z@W + b); scatter into Sbd ----------------
__global__ void assign_kernel(const float* __restrict__ Z, const float* __restrict__ W,
                              const float* __restrict__ bias, float* __restrict__ out) {
  __shared__ float zrow[8][Fdim];
  int wave = threadIdx.x >> 5, lane = threadIdx.x & 31;
  int r = blockIdx.x * 8 + wave;          // node index
  #pragma unroll
  for (int j = 0; j < 4; ++j) zrow[wave][lane + 32*j] = Z[(size_t)r * Fdim + lane + 32*j];
  __syncthreads();
  float acc0 = 0.f, acc1 = 0.f;
  #pragma unroll 4
  for (int f = 0; f < Fdim; ++f) {
    float zv = zrow[wave][f];
    acc0 += zv * W[f*Kc + lane];
    acc1 += zv * W[f*Kc + lane + 32];
  }
  acc0 += bias[lane]; acc1 += bias[lane + 32];
  float mx = fmaxf(acc0, acc1);
  #pragma unroll
  for (int off = 16; off; off >>= 1) mx = fmaxf(mx, __shfl_xor(mx, off, 32));
  float e0 = __expf(acc0 - mx), e1 = __expf(acc1 - mx);
  float sm = e0 + e1;
  #pragma unroll
  for (int off = 16; off; off >>= 1) sm += __shfl_xor(sm, off, 32);
  float inv = 1.0f / sm;
  float s0 = e0 * inv, s1 = e1 * inv;
  float* SL = out + SL_OFF;
  SL[(size_t)r * Kc + lane]      = s0;
  SL[(size_t)r * Kc + lane + 32] = s1;
  int g = r >> 9;                          // node -> graph (512 nodes/graph)
  float* SBD = out + SBD_OFF;
  SBD[(size_t)r * KB_TOT + g*Kc + lane]      = s0;
  SBD[(size_t)r * KB_TOT + g*Kc + lane + 32] = s1;
}

// ---------------- kernel 3: Zp[g] = S_g.T @ Z_g  (64x128, K=512) ----------------
// LDS: SgT[64][LDK] f16 + ZT[128][LDK] f16 (~195 KB, needs CDNA5's 320 KB/WGP)
__global__ void zp_kernel(const float* __restrict__ SL, const float* __restrict__ Z,
                          float* __restrict__ out) {
  extern __shared__ _Float16 smem[];
  _Float16* SgT = smem;                    // [Kc][LDK]
  _Float16* ZT  = smem + (size_t)Kc * LDK; // [Fdim][LDK]
  int g = blockIdx.x;
  const float* Ssrc = SL + (size_t)g * NPG * Kc;
  const float* Zg   = Z  + (size_t)g * NPG * Fdim;
  for (int i = threadIdx.x; i < NPG * Kc; i += 256) {
    int node = i >> 6, c = i & (Kc - 1);
    SgT[(size_t)c * LDK + node] = (_Float16)Ssrc[i];
  }
  for (int i = threadIdx.x; i < NPG * Fdim; i += 256) {
    int node = i >> 7, f = i & (Fdim - 1);
    ZT[(size_t)f * LDK + node] = (_Float16)Zg[i];
  }
  __syncthreads();
  int wave = threadIdx.x >> 5;
  int mt = wave & 3;                       // 16-row cluster tile
  int ng = (wave >> 2) * 4;                // 4 feature tiles of 16
  v8f c[4];
  #pragma unroll
  for (int j = 0; j < 4; ++j) c[j] = (v8f){0.f,0.f,0.f,0.f,0.f,0.f,0.f,0.f};
  for (int kt = 0; kt < NPG/32; ++kt) {
    v16h a = load_at_frag_ldsT(SgT, mt*16, kt*32);     // (S_g.T)[m][k]
    #pragma unroll
    for (int j = 0; j < 4; ++j) {
      v16h bf = load_b_frag_ldsT(ZT, kt*32, (ng + j)*16);
      c[j] = __builtin_amdgcn_wmma_f32_16x16x32_f16(false, a, false, bf,
                                                    (short)0, c[j], false, false);
    }
  }
  float* Zp = out + ZP_OFF;
  #pragma unroll
  for (int j = 0; j < 4; ++j)
    store_c_frag_global(Zp, g*Kc + mt*16, (ng + j)*16, Fdim, c[j]);
}

// ---------------- kernel 4: Ap[p,q] = S_p.T @ A[p-blk, q-blk] @ S_q ----------------
// One block per (p,q) pair (1024 blocks). LDS: SqT + SpT + TT, each [64][LDK] f16 (~195 KB).
// Phase 1: T = A_blk @ S_q (512x64, K=512) — streams the 1 GiB A exactly once (HBM roofline).
// Phase 2: Ap_blk = S_p.T @ T (64x64, K=512).
__global__ void ap_kernel(const float* __restrict__ SL, const float* __restrict__ A,
                          float* __restrict__ out) {
  extern __shared__ _Float16 smem[];
  _Float16* SqT = smem;                        // [Kc][LDK]
  _Float16* SpT = smem + (size_t)Kc * LDK;     // [Kc][LDK]
  _Float16* TT  = smem + (size_t)2 * Kc * LDK; // [Kc][LDK]  (T[k][n] stored as TT[n][k])
  int p = blockIdx.x >> 5, q = blockIdx.x & 31;
  const float* Sq_g = SL + (size_t)q * NPG * Kc;
  const float* Sp_g = SL + (size_t)p * NPG * Kc;
  for (int i = threadIdx.x; i < NPG * Kc; i += 256) {
    int node = i >> 6, c = i & (Kc - 1);
    SqT[(size_t)c * LDK + node] = (_Float16)Sq_g[i];
    SpT[(size_t)c * LDK + node] = (_Float16)Sp_g[i];
  }
  __syncthreads();

  int wave = threadIdx.x >> 5;
  int lane = threadIdx.x & 31;
  const float* Ablk = A + (size_t)(p * NPG) * N_TOT + (size_t)q * NPG;

  // ---- phase 1: each wave owns 64 rows of T ----
  int r0 = wave * 64;
  v8f c[4][4];
  #pragma unroll
  for (int m = 0; m < 4; ++m)
    #pragma unroll
    for (int n = 0; n < 4; ++n) c[m][n] = (v8f){0.f,0.f,0.f,0.f,0.f,0.f,0.f,0.f};
  for (int kt = 0; kt < NPG/32; ++kt) {
    // prefetch the A k-tile 3 steps ahead (one 128B line == one row's 32-float k-tile)
    if (kt + 3 < NPG/32) {
      __builtin_prefetch(Ablk + (size_t)(r0 + lane)      * N_TOT + (kt + 3)*32, 0, 0);
      __builtin_prefetch(Ablk + (size_t)(r0 + lane + 32) * N_TOT + (kt + 3)*32, 0, 0);
    }
    v16h bf[4];
    #pragma unroll
    for (int n = 0; n < 4; ++n) bf[n] = load_b_frag_ldsT(SqT, kt*32, n*16);
    #pragma unroll
    for (int m = 0; m < 4; ++m) {
      v16h a = load_a_frag_global(Ablk, r0 + m*16, kt*32, N_TOT);
      #pragma unroll
      for (int n = 0; n < 4; ++n)
        c[m][n] = __builtin_amdgcn_wmma_f32_16x16x32_f16(false, a, false, bf[n],
                                                         (short)0, c[m][n], false, false);
    }
  }
  #pragma unroll
  for (int m = 0; m < 4; ++m)
    #pragma unroll
    for (int n = 0; n < 4; ++n)
      store_c_frag_ldsT(TT, r0 + m*16, n*16, c[m][n]);
  __syncthreads();

  // ---- phase 2: 16 output tiles, 2 per wave ----
  int mt = wave >> 1;
  int nb = (wave & 1) * 2;
  v8f d[2];
  d[0] = (v8f){0.f,0.f,0.f,0.f,0.f,0.f,0.f,0.f};
  d[1] = d[0];
  for (int kt = 0; kt < NPG/32; ++kt) {
    v16h a = load_at_frag_ldsT(SpT, mt*16, kt*32);     // (S_p.T)[m][k]
    #pragma unroll
    for (int j = 0; j < 2; ++j) {
      v16h bf = load_b_frag_ldsT(TT, kt*32, (nb + j)*16);
      d[j] = __builtin_amdgcn_wmma_f32_16x16x32_f16(false, a, false, bf,
                                                    (short)0, d[j], false, false);
    }
  }
  float* Ap = out + AP_OFF;
  #pragma unroll
  for (int j = 0; j < 2; ++j)
    store_c_frag_global(Ap, p*Kc + mt*16, q*Kc + (nb + j)*16, KB_TOT, d[j]);
}

// ---------------- launch ----------------
extern "C" void kernel_launch(void* const* d_in, const int* in_sizes, int n_in,
                              void* d_out, int out_size, void* d_ws, size_t ws_size,
                              hipStream_t stream) {
  const float* Z    = (const float*)d_in[0];
  const float* A    = (const float*)d_in[1];
  // d_in[2] = n_nodes (all 512 in this packed workload)
  const float* W    = (const float*)d_in[3];
  const float* bias = (const float*)d_in[4];
  float* out = (float*)d_out;
  const float* SLp = out + SL_OFF;

  size_t lds_zp = (size_t)(Kc + Fdim) * LDK * 2;   // f16 bytes
  size_t lds_ap = (size_t)3 * Kc * LDK * 2;

  fill_kernel  <<<32768, 256, 0, stream>>>(out);
  assign_kernel<<<N_TOT/8, 256, 0, stream>>>(Z, W, bias, out);
  zp_kernel    <<<Bg, 256, lds_zp, stream>>>(SLp, Z, out);
  ap_kernel    <<<Bg*Bg, 256, lds_ap, stream>>>(SLp, A, out);
}
